// Attention_29351806500887
// MI455X (gfx1250) — matrix operand order repaired and checked by reading, not compiled, not run on hardware
//
#include <hip/hip_runtime.h>
#include <stdint.h>
#include <math.h>

typedef __attribute__((ext_vector_type(16))) __bf16 v16bf;
typedef __attribute__((ext_vector_type(8)))  __bf16 v8bf;
typedef __attribute__((ext_vector_type(8)))  float  v8f;
typedef uint32_t u32x4 __attribute__((ext_vector_type(4)));
typedef uint32_t u32x8 __attribute__((ext_vector_type(8)));

#define DIMC   768
#define HEADS  12
#define DH     64
#define BATCH  8
#define SEQ    1024
#define NTOK   (BATCH * SEQ)      // 8192
#define QKV_N  (3 * DIMC)         // 2304

// ---------------------------------------------------------------------------
// helpers
// ---------------------------------------------------------------------------
static __device__ inline v16bf frag_combine(const __bf16* lo, const __bf16* hi) {
  v8bf a = *(const v8bf*)lo;
  v8bf b = *(const v8bf*)hi;
  v16bf r;
#pragma unroll
  for (int i = 0; i < 8; ++i) { r[i] = a[i]; r[i + 8] = b[i]; }
  return r;
}

static __device__ inline v8f wmma_bf16(v16bf a, v16bf b, v8f c) {
  // D = A(16x32 bf16) * B(32x16 bf16) + C(16x16 f32)
  return __builtin_amdgcn_wmma_f32_16x16x32_bf16(
      /*neg_a=*/false, a, /*neg_b=*/false, b,
      /*c_mod=*/(short)0, c, /*reuse_a=*/false, /*reuse_b=*/false);
}

static __device__ inline v8f vzero8() {
  v8f z;
#pragma unroll
  for (int i = 0; i < 8; ++i) z[i] = 0.0f;
  return z;
}

// xor-butterfly exchange via ds_swizzle_b32 (group-of-32 mode, immediate
// pattern: and_mask=0x1F, or_mask=0, xor_mask=N). Masks < 16 never cross the
// 16-lane halves, so the per-half row reductions stay independent.
#define SWZ_XOR(v, n) \
  __int_as_float(__builtin_amdgcn_ds_swizzle(__float_as_int(v), ((n) << 10) | 0x1F))

static __device__ inline float bfly_max16(float v) {
  v = fmaxf(v, SWZ_XOR(v, 1));
  v = fmaxf(v, SWZ_XOR(v, 2));
  v = fmaxf(v, SWZ_XOR(v, 4));
  v = fmaxf(v, SWZ_XOR(v, 8));
  return v;
}
static __device__ inline float bfly_sum16(float v) {
  v += SWZ_XOR(v, 1);
  v += SWZ_XOR(v, 2);
  v += SWZ_XOR(v, 4);
  v += SWZ_XOR(v, 8);
  return v;
}

// LDS byte offset of a __shared__ object: generic pointers to LDS carry the
// workgroup-relative offset in their low 32 bits (ISA 10.2 aperture mapping).
static __device__ inline uint32_t lds_off(const void* p) {
  return (uint32_t)(uintptr_t)p;
}

// Issue a 2D TDM load: copy a (height x width) bf16 tile with row stride
// `stride_elems` from global memory into LDS (row-major, densely packed).
// D# layout per CDNA5 ISA ch.8 (group0 128b + group1 256b, groups 2/3 = NULL).
static __device__ inline void tdm_load_2d(uint32_t lds_byte_addr,
                                          const void* gptr,
                                          uint32_t width_elems,
                                          uint32_t height_rows,
                                          uint32_t stride_elems) {
  uint64_t ga = (uint64_t)(uintptr_t)gptr;
  u32x4 g0;
  g0[0] = 1u;                                               // count=1 (user D#)
  g0[1] = lds_byte_addr;                                    // lds_addr
  g0[2] = (uint32_t)ga;                                     // global_addr[31:0]
  g0[3] = (uint32_t)((ga >> 32) & 0x1FFFFFFu) | (2u << 30); // addr[56:32] | type=2
  u32x8 g1;
  g1[0] = (1u << 16);                                       // data_size=1 (2B)
  g1[1] = (width_elems & 0xFFFFu) << 16;                    // tensor_dim0[15:0]
  g1[2] = (width_elems >> 16) | ((height_rows & 0xFFFFu) << 16); // d0 hi | d1 lo
  g1[3] = (height_rows >> 16) | ((width_elems & 0xFFFFu) << 16); // d1 hi | tile_dim0
  g1[4] = height_rows & 0xFFFFu;                            // tile_dim1 | tile_dim2=0
  g1[5] = stride_elems;                                     // dim0_stride[31:0]
  g1[6] = 0;                                                // dim0_stride hi | dim1_stride lo
  g1[7] = 0;
  asm volatile("tensor_load_to_lds %0, %1" :: "s"(g0), "s"(g1) : "memory");
}

// ---------------------------------------------------------------------------
// f32 -> bf16 convert; first `scaleElems` elements get *scale (Q rows of wqkv
// pick up 0.125 * log2(e) so flash softmax can run in the exp2 domain).
// ---------------------------------------------------------------------------
__global__ void cvt_bf16_kernel(const float* __restrict__ in,
                                __bf16* __restrict__ out,
                                int n, int scaleElems, float scale) {
  int i = blockIdx.x * blockDim.x + threadIdx.x;
  int stride = gridDim.x * blockDim.x;
  for (; i < n; i += stride) {
    float v = in[i];
    if (i < scaleElems) v *= scale;
    out[i] = (__bf16)v;
  }
}

// ---------------------------------------------------------------------------
// C[M x N] = A[M x 768](bf16, row-major) * W[N x 768](bf16, row-major)^T
// Block tile 128x128, 8 waves (2x4), wave tile 64x32, K-step 32.
// Tiles delivered by TDM into double-buffered LDS; wave 0 drives the DMA.
// ---------------------------------------------------------------------------
template <bool OUT_BF16>
__global__ __launch_bounds__(256) void gemm_wmma_kernel(
    const __bf16* __restrict__ A, const __bf16* __restrict__ W,
    __bf16* __restrict__ Cb, float* __restrict__ Cf,
    const float* __restrict__ bias, int ldc) {
  __shared__ alignas(16) __bf16 As[2][128][32];
  __shared__ alignas(16) __bf16 Bs[2][128][32];

  const int tid  = threadIdx.x;
  const int lane = tid & 31;
  const int wave = tid >> 5;
  const int wm   = wave >> 2;   // 0..1
  const int wn   = wave & 3;    // 0..3
  const int bm   = blockIdx.x * 128;
  const int bn   = blockIdx.y * 128;
  const int half = lane >> 4;   // 0 / 1
  const int l16  = lane & 15;
  const bool issuer = (wave == 0);

  const uint32_t asB = lds_off(&As[0][0][0]);
  const uint32_t bsB = lds_off(&Bs[0][0][0]);
  const uint32_t bufBytes = 128 * 32 * 2;  // 8 KB

  v8f acc[4][2];
#pragma unroll
  for (int sm = 0; sm < 4; ++sm)
#pragma unroll
    for (int sn = 0; sn < 2; ++sn) acc[sm][sn] = vzero8();

  // prologue: DMA first K-tiles into buffer 0
  if (issuer) {
    tdm_load_2d(asB, A + (size_t)bm * DIMC, 32, 128, DIMC);
    tdm_load_2d(bsB, W + (size_t)bn * DIMC, 32, 128, DIMC);
  }

  int buf = 0;
  for (int k0 = 0; k0 < DIMC; k0 += 32, buf ^= 1) {
    const int nk = k0 + 32;
    if (issuer) {
      if (nk < DIMC) {
        // queue next pair into the other buffer, then wait for current pair
        tdm_load_2d(asB + (buf ^ 1) * bufBytes, A + (size_t)bm * DIMC + nk, 32, 128, DIMC);
        tdm_load_2d(bsB + (buf ^ 1) * bufBytes, W + (size_t)bn * DIMC + nk, 32, 128, DIMC);
        __builtin_amdgcn_s_wait_tensorcnt(2);
      } else {
        __builtin_amdgcn_s_wait_tensorcnt(0);
      }
    }
    __syncthreads();   // publish current tiles to all waves

    // B fragments: element (k,n) = Bs[n][k]; lane n=l16, k = half*8 + {0..7, 16..23}
    v16bf bf[2];
#pragma unroll
    for (int sn = 0; sn < 2; ++sn) {
      const __bf16* p = &Bs[buf][wn * 32 + sn * 16 + l16][half * 8];
      bf[sn] = frag_combine(p, p + 16);
    }
#pragma unroll
    for (int sm = 0; sm < 4; ++sm) {
      // A fragment: lane row m, halves: K = half*8 + {0..7, 16..23}
      const __bf16* pa = &As[buf][wm * 64 + sm * 16 + l16][half * 8];
      v16bf af = frag_combine(pa, pa + 16);
#pragma unroll
      for (int sn = 0; sn < 2; ++sn)
        acc[sm][sn] = wmma_bf16(af, bf[sn], acc[sm][sn]);
    }
    __syncthreads();   // all waves done with this buffer before DMA reuses it
  }

  // C/D layout: lanes 0-15 -> rows r(0..7), lanes 16-31 -> rows r+8; col = l16
#pragma unroll
  for (int sm = 0; sm < 4; ++sm) {
#pragma unroll
    for (int sn = 0; sn < 2; ++sn) {
      int gm = bm + wm * 64 + sm * 16 + half * 8;
      int gn = bn + wn * 32 + sn * 16 + l16;
      float bv = OUT_BF16 ? 0.0f : bias[gn];
#pragma unroll
      for (int r = 0; r < 8; ++r) {
        float v = acc[sm][sn][r];
        if (OUT_BF16) Cb[(size_t)(gm + r) * ldc + gn] = (__bf16)v;
        else          Cf[(size_t)(gm + r) * ldc + gn] = v + bv;
      }
    }
  }
}

// ---------------------------------------------------------------------------
// Flash attention. qkv: [8192, 2304] bf16 (Q pre-scaled by 0.125*log2e, so
// softmax runs in the exp2 domain: one v_exp_f32 per probability).
// Block: 128 threads (4 waves) handles 64 query rows for one (b,h).
// K/V tiles delivered by TDM, double-buffered; V transposed LDS->LDS.
// ---------------------------------------------------------------------------
__global__ __launch_bounds__(128) void flash_attn_kernel(
    const __bf16* __restrict__ qkv, __bf16* __restrict__ out) {
  __shared__ alignas(16) __bf16 Qs[64][64];
  __shared__ alignas(16) __bf16 Ks[2][32][64];
  __shared__ alignas(16) __bf16 Vs[2][32][64];
  __shared__ alignas(16) __bf16 VTs[64][32];
  __shared__ alignas(16) __bf16 Ps[4][16][32];

  const int tid  = threadIdx.x;
  const int lane = tid & 31;
  const int wave = tid >> 5;  // 0..3
  const int half = lane >> 4;
  const int l16  = lane & 15;
  const bool issuer = (wave == 0);

  const int qt = blockIdx.x;  // query tile (0..15)
  const int h  = blockIdx.y;  // head
  const int b  = blockIdx.z;  // batch
  const int tok0 = b * SEQ + qt * 64;
  const __bf16* kBase = qkv + (size_t)(b * SEQ) * QKV_N + DIMC + h * DH;
  const __bf16* vBase = qkv + (size_t)(b * SEQ) * QKV_N + 2 * DIMC + h * DH;

  // prologue DMA: Q tile + first K/V pair into buffer 0
  if (issuer) {
    tdm_load_2d(lds_off(&Qs[0][0]),
                qkv + (size_t)tok0 * QKV_N + h * DH, DH, 64, QKV_N);
    tdm_load_2d(lds_off(&Ks[0][0][0]), kBase, DH, 32, QKV_N);
    tdm_load_2d(lds_off(&Vs[0][0][0]), vBase, DH, 32, QKV_N);
    __builtin_amdgcn_s_wait_tensorcnt(0);
  }
  __syncthreads();

  // Q A-fragments (this wave's 16 rows), K-dim 64 split into two 32-chunks
  v16bf qf[2];
  {
    const __bf16* p0 = &Qs[wave * 16 + l16][half * 8];
    qf[0] = frag_combine(p0, p0 + 16);
    const __bf16* p1 = &Qs[wave * 16 + l16][32 + half * 8];
    qf[1] = frag_combine(p1, p1 + 16);
  }

  v8f o[4];
#pragma unroll
  for (int t = 0; t < 4; ++t) o[t] = vzero8();
  float mrow[8], lrow[8];
#pragma unroll
  for (int r = 0; r < 8; ++r) { mrow[r] = -1.0e30f; lrow[r] = 0.0f; }

  int buf = 0;
  for (int kt = 0; kt < SEQ; kt += 32, buf ^= 1) {
    const int nkt = kt + 32;
    if (issuer) {
      if (nkt < SEQ) {
        // queue next K/V pair into the other buffer, wait for current pair
        tdm_load_2d(lds_off(&Ks[buf ^ 1][0][0]), kBase + (size_t)nkt * QKV_N, DH, 32, QKV_N);
        tdm_load_2d(lds_off(&Vs[buf ^ 1][0][0]), vBase + (size_t)nkt * QKV_N, DH, 32, QKV_N);
        __builtin_amdgcn_s_wait_tensorcnt(2);
      } else {
        __builtin_amdgcn_s_wait_tensorcnt(0);
      }
    }
    __syncthreads();   // current K/V visible to all waves

    // transpose V in LDS: VTs[d][k] = Vs[buf][k][d]
#pragma unroll
    for (int t = 0; t < 8; ++t) {
      int idx = tid + t * 128;
      int k   = idx >> 5;   // 0..31
      int d2  = idx & 31;   // d = 2*d2
      uint32_t v = *(const uint32_t*)&Vs[buf][k][2 * d2];
      __bf16 two[2];
      *(uint32_t*)two = v;
      VTs[2 * d2][k]     = two[0];
      VTs[2 * d2 + 1][k] = two[1];
    }
    __syncthreads();

    // S = Q * K^T for two 16-key sub-tiles (chained WMMAs over K-dim 64)
    v8f sf[2];
#pragma unroll
    for (int sn = 0; sn < 2; ++sn) {
      const __bf16* pb0 = &Ks[buf][sn * 16 + l16][half * 8];
      v16bf kb0 = frag_combine(pb0, pb0 + 16);
      const __bf16* pb1 = &Ks[buf][sn * 16 + l16][32 + half * 8];
      v16bf kb1 = frag_combine(pb1, pb1 + 16);
      v8f s0 = wmma_bf16(qf[0], kb0, vzero8());
      sf[sn] = wmma_bf16(qf[1], kb1, s0);
    }

    // online softmax (exp2 domain): row r of this lane-half
#pragma unroll
    for (int r = 0; r < 8; ++r) {
      float v = bfly_max16(fmaxf(sf[0][r], sf[1][r]));
      float mnew  = fmaxf(mrow[r], v);
      float alpha = exp2f(mrow[r] - mnew);
      mrow[r] = mnew;
      float p0 = exp2f(sf[0][r] - mnew);
      float p1 = exp2f(sf[1][r] - mnew);
      float rs = bfly_sum16(p0 + p1);
      lrow[r] = lrow[r] * alpha + rs;
      // stage P (C-layout -> LDS matrix form)
      Ps[wave][half * 8 + r][l16]      = (__bf16)p0;
      Ps[wave][half * 8 + r][16 + l16] = (__bf16)p1;
      // rescale running output
#pragma unroll
      for (int t = 0; t < 4; ++t) o[t][r] *= alpha;
    }
    __builtin_amdgcn_wave_barrier();

    // reload P as A-fragment (16x32)
    const __bf16* pp = &Ps[wave][l16][half * 8];
    v16bf pf = frag_combine(pp, pp + 16);

  // O += P * V ; B-frag element (k,n) = VTs[dh][k] (k = half*8 + {0..7, 16..23})
#pragma unroll
    for (int t = 0; t < 4; ++t) {
      const __bf16* pv = &VTs[t * 16 + l16][half * 8];
      v16bf vf = frag_combine(pv, pv + 16);
      o[t] = wmma_bf16(pf, vf, o[t]);
    }
    __syncthreads();   // done with this buffer before DMA refills it
  }

  // finalize: divide by l, write [tok, h*64 + d] bf16
#pragma unroll
  for (int r = 0; r < 8; ++r) {
    float inv = 1.0f / lrow[r];
    int tok = tok0 + wave * 16 + half * 8 + r;
#pragma unroll
    for (int t = 0; t < 4; ++t)
      out[(size_t)tok * DIMC + h * DH + t * 16 + l16] = (__bf16)(o[t][r] * inv);
  }
}

// ---------------------------------------------------------------------------
// launch
// ---------------------------------------------------------------------------
extern "C" void kernel_launch(void* const* d_in, const int* in_sizes, int n_in,
                              void* d_out, int out_size, void* d_ws, size_t ws_size,
                              hipStream_t stream) {
  const float* x      = (const float*)d_in[0];
  const float* w_qkv  = (const float*)d_in[1];
  const float* w_proj = (const float*)d_in[2];
  const float* b_proj = (const float*)d_in[3];
  float* outf = (float*)d_out;

  __bf16* ws     = (__bf16*)d_ws;
  __bf16* xb     = ws;                                   // 8192*768
  __bf16* wqkvb  = xb + (size_t)NTOK * DIMC;             // 2304*768
  __bf16* wprojb = wqkvb + (size_t)QKV_N * DIMC;         // 768*768
  __bf16* qkvb   = wprojb + (size_t)DIMC * DIMC;         // 8192*2304
  __bf16* attnb  = qkvb + (size_t)NTOK * QKV_N;          // 8192*768

  // converts; Q rows of w_qkv pick up softmax scale * log2(e)
  const float qscale = 0.125f * 1.44269504088896340736f;
  cvt_bf16_kernel<<<1024, 256, 0, stream>>>(x, xb, NTOK * DIMC, 0, 1.0f);
  cvt_bf16_kernel<<<512, 256, 0, stream>>>(w_qkv, wqkvb, QKV_N * DIMC, DIMC * DIMC, qscale);
  cvt_bf16_kernel<<<256, 256, 0, stream>>>(w_proj, wprojb, DIMC * DIMC, 0, 1.0f);

  // qkv = x @ w_qkv^T  (bf16 out)
  gemm_wmma_kernel<true><<<dim3(NTOK / 128, QKV_N / 128), 256, 0, stream>>>(
      xb, wqkvb, qkvb, nullptr, nullptr, QKV_N);

  // fused attention per (b, h, 64-query tile)
  flash_attn_kernel<<<dim3(SEQ / 64, HEADS, BATCH), 128, 0, stream>>>(qkvb, attnb);

  // out = attn @ w_proj^T + b_proj  (f32 out)
  gemm_wmma_kernel<false><<<dim3(NTOK / 128, DIMC / 128), 256, 0, stream>>>(
      attnb, wprojb, nullptr, outf, b_proj, DIMC);
}
